// GCN_13657996001620
// MI455X (gfx1250) — matrix operand order repaired and checked by reading, not compiled, not run on hardware
//
#include <hip/hip_runtime.h>
#include <hip/hip_bf16.h>

// ---------------------------------------------------------------------------
// GCN forward for gfx1250 (MI455X). SpMM via L2 float atomics; dense GEMMs via
// WMMA. Prefers full-precision V_WMMA_F32_16X16X4_F32 when the builtin exists;
// falls back to codegen-confirmed V_WMMA_F32_16X16X32_F16 (f32 accumulate).
// ---------------------------------------------------------------------------

typedef __attribute__((ext_vector_type(2)))  float    v2f;
typedef __attribute__((ext_vector_type(8)))  float    v8f;
typedef __attribute__((ext_vector_type(16))) _Float16 v16h;

#define K_DIM 128
#define N_NODES 10000
#define N_EDGES 640000

#if defined(__gfx1250__) && __has_builtin(__builtin_amdgcn_wmma_f32_16x16x4_f32)
#define USE_F32_WMMA 1
#else
#define USE_F32_WMMA 0
#endif

__device__ __forceinline__ void atomicAddF(float* p, float v) {
  __hip_atomic_fetch_add(p, v, __ATOMIC_RELAXED, __HIP_MEMORY_SCOPE_AGENT);
}

// ------------------------- utility: zero a float buffer ---------------------
__global__ void gcn_zero_kernel(float* __restrict__ p, int n) {
  int t = blockIdx.x * blockDim.x + threadIdx.x;
  if (t < n) p[t] = 0.0f;
}

// ------------------- weight prep: W[K x N] -> Wt[N x K] f16 -----------------
__global__ void gcn_prep_w_kernel(const float* __restrict__ W,
                                  _Float16* __restrict__ Wt, int N) {
  int t = blockIdx.x * blockDim.x + threadIdx.x;  // over N*K
  if (t >= N * K_DIM) return;
  int nn = t / K_DIM, kk = t % K_DIM;
  Wt[t] = (_Float16)W[kk * N + nn];
}

// ----- W3 prep: pad 128x40 -> 128x48 (f32 row-major) + 48x128 f16 transposed
__global__ void gcn_prep_w3_kernel(const float* __restrict__ W3,
                                   const float* __restrict__ b3,
                                   float* __restrict__ W3p,
                                   float* __restrict__ b3p,
                                   _Float16* __restrict__ Wt3) {
  int t = blockIdx.x * blockDim.x + threadIdx.x;  // over 48*K_DIM
  if (t < 48) b3p[t] = (t < 40) ? b3[t] : 0.0f;
  if (t >= 48 * K_DIM) return;
  int nn = t / K_DIM, kk = t % K_DIM;
  float w = (nn < 40) ? W3[kk * 40 + nn] : 0.0f;
  Wt3[t] = (_Float16)w;
  W3p[kk * 48 + nn] = w;
}

// ---------------------------- SpMM: y += A_sparse * x -----------------------
// One wave per edge; lane handles 4 consecutive features (float4 gather).
__global__ void gcn_spmm_kernel(const int* __restrict__ row,
                                const int* __restrict__ col,
                                const float* __restrict__ val,
                                const float* __restrict__ x,
                                float* __restrict__ y, int E) {
  int t = blockIdx.x * blockDim.x + threadIdx.x;
  int e = t >> 5;
  int lane = t & 31;
  if (e >= E) return;
  int r = row[e], c = col[e];
  float v = val[e];
  const float4 xv = *(const float4*)(x + (size_t)c * K_DIM + lane * 4);
  float* yp = y + (size_t)r * K_DIM + lane * 4;
  atomicAddF(yp + 0, v * xv.x);
  atomicAddF(yp + 1, v * xv.y);
  atomicAddF(yp + 2, v * xv.z);
  atomicAddF(yp + 3, v * xv.w);
}

// -------------------- WMMA GEMM: C = relu?(A[MxK] * W[KxN] + b) -------------
// grid.x = M/16 (row strips), one wave per 16-col tile (blockDim = 32*N/16).
__global__ void gcn_gemm_wmma_kernel(const float* __restrict__ A,
                                     const float* __restrict__ Wf,   // K x N f32
                                     const _Float16* __restrict__ Wt,// N x K f16
                                     const float* __restrict__ bias,
                                     float* __restrict__ C, int N, int do_relu) {
  const int lane  = threadIdx.x & 31;
  const int wave  = threadIdx.x >> 5;
  const int mbase = blockIdx.x * 16;
  const int nbase = wave * 16;
  const int half  = lane >> 4;   // which 16-lane half of the wave
  const int l15   = lane & 15;
  const int m     = mbase + l15;
  const int n     = nbase + l15;

  v8f acc = {};
#if USE_F32_WMMA
  // A 16x4 fp32 fragment: lane(m=l15); VGPR0/1 hold K = 2*half, 2*half+1.
  // B 4x16 fp32 fragment: mirrored -> rows K = 2*half, 2*half+1 at column n.
  for (int k = 0; k < K_DIM; k += 4) {
    v2f a, b;
    const float* ap = A + (size_t)m * K_DIM + k + 2 * half;
    a[0] = ap[0];
    a[1] = ap[1];
    const int kk = k + 2 * half;
    b[0] = Wf[(size_t)kk * N + n];
    b[1] = Wf[(size_t)(kk + 1) * N + n];
    acc = __builtin_amdgcn_wmma_f32_16x16x4_f32(false, a, false, b,
                                                (short)0, acc, false, false);
  }
#else
  (void)Wf;
  // A 16x32 f16: elems 0..7 -> K = k + 8*half + j ; elems 8..15 -> +16.
  // B 32x16 f16: elems 0..7 -> K = k + 8*half + j ; elems 8..15 -> +16, at column n (Wt is N x K).
  for (int k = 0; k < K_DIM; k += 32) {
    v16h a, b;
    const float* ap = A + (size_t)m * K_DIM + k + half * 8;
#pragma unroll
    for (int j = 0; j < 8; ++j) a[j] = (_Float16)ap[j];
#pragma unroll
    for (int j = 0; j < 8; ++j) a[8 + j] = (_Float16)ap[16 + j];
    const _Float16* bp = Wt + (size_t)n * K_DIM + k + half * 8;
#pragma unroll
    for (int j = 0; j < 8; ++j) b[j] = bp[j];
#pragma unroll
    for (int j = 0; j < 8; ++j) b[8 + j] = bp[16 + j];
    acc = __builtin_amdgcn_wmma_f32_16x16x32_f16(false, a, false, b,
                                                 (short)0, acc, false, false);
  }
#endif
  // C/D layout: element v of acc -> row mbase + v + 8*half, col nbase + l15.
  const float bv = bias[n];
#pragma unroll
  for (int v = 0; v < 8; ++v) {
    int rrow = mbase + v + 8 * half;
    float out = acc[v] + bv;
    if (do_relu) out = fmaxf(out, 0.0f);
    C[(size_t)rrow * N + n] = out;
  }
}

// ------------------- BatchNorm pass 1: per-channel sum / sumsq --------------
__global__ void gcn_bn_stats_kernel(const float* __restrict__ H,
                                    float* __restrict__ stats, int M) {
  int c = threadIdx.x;            // 0..127 (channel)
  int r0 = blockIdx.x * 64;
  int rend = r0 + 64; if (rend > M) rend = M;
  float s = 0.0f, sq = 0.0f;
  for (int r = r0; r < rend; ++r) {
    float v = H[(size_t)r * K_DIM + c];
    s += v;
    sq += v * v;
  }
  atomicAddF(&stats[c], s);
  atomicAddF(&stats[K_DIM + c], sq);
}

// ------------------- BatchNorm pass 2: normalize (biased var) ---------------
__global__ void gcn_bn_norm_kernel(const float* __restrict__ H,
                                   const float* __restrict__ stats,
                                   const float* __restrict__ gamma,
                                   const float* __restrict__ beta,
                                   float* __restrict__ O, int M) {
  int t = blockIdx.x * blockDim.x + threadIdx.x;
  if (t >= M * K_DIM) return;
  int c = t & (K_DIM - 1);
  const float invM = 1.0f / (float)M;
  float mean = stats[c] * invM;
  float var  = stats[K_DIM + c] * invM - mean * mean;
  float rstd = rsqrtf(var + 1e-5f);
  O[t] = gamma[c] * (H[t] - mean) * rstd + beta[c];
}

// ---------------------- log_softmax over first 40 of 48 ---------------------
__global__ void gcn_logsoftmax_kernel(const float* __restrict__ L,
                                      float* __restrict__ out, int M) {
  int r = blockIdx.x * blockDim.x + threadIdx.x;
  if (r >= M) return;
  const float* lp = L + (size_t)r * 48;
  float mx = -3.402823466e38f;
  for (int j = 0; j < 40; ++j) mx = fmaxf(mx, lp[j]);
  float s = 0.0f;
  for (int j = 0; j < 40; ++j) s += expf(lp[j] - mx);
  float lse = logf(s) + mx;
  for (int j = 0; j < 40; ++j) out[(size_t)r * 40 + j] = lp[j] - lse;
}

// ---------------------------------------------------------------------------
extern "C" void kernel_launch(void* const* d_in, const int* in_sizes, int n_in,
                              void* d_out, int out_size, void* d_ws, size_t ws_size,
                              hipStream_t stream) {
  (void)in_sizes; (void)n_in; (void)out_size; (void)ws_size;
  const float* x      = (const float*)d_in[0];
  const int*   erow   = (const int*)d_in[1];
  const int*   ecol   = (const int*)d_in[2];
  const float* eval   = (const float*)d_in[3];
  const float* W1     = (const float*)d_in[4];
  const float* b1     = (const float*)d_in[5];
  const float* gamma2 = (const float*)d_in[6];
  const float* beta2  = (const float*)d_in[7];
  const float* W2     = (const float*)d_in[8];
  const float* b2     = (const float*)d_in[9];
  const float* gamma3 = (const float*)d_in[10];
  const float* beta3  = (const float*)d_in[11];
  const float* W3     = (const float*)d_in[12];
  const float* b3     = (const float*)d_in[13];
  float* out = (float*)d_out;

  // ---- workspace carve-up (floats) ----
  float* ws     = (float*)d_ws;
  float* hA     = ws;                      // 10000*128 = 1,280,000
  float* hB     = ws + 1280000;            // 1,280,000
  float* logits = ws + 2560000;            // 10000*48 = 480,000
  float* stats  = ws + 3040000;            // 256
  float* W3p    = ws + 3040256;            // 128*48 = 6144
  float* b3p    = ws + 3046400;            // 48 (+16 align pad)
  _Float16* Wt1 = (_Float16*)(ws + 3046464);  // 128*128 halves
  _Float16* Wt2 = Wt1 + 16384;
  _Float16* Wt3 = Wt2 + 16384;             // 48*128 halves

  const int M = N_NODES, E = N_EDGES;
  const int elemsH   = M * K_DIM;          // 1,280,000
  const int spmm_thr = E * 32;

  // ---- weight prep (both f32-padded and f16-transposed forms) ----
  gcn_prep_w_kernel<<<(K_DIM * K_DIM + 255) / 256, 256, 0, stream>>>(W1, Wt1, K_DIM);
  gcn_prep_w_kernel<<<(K_DIM * K_DIM + 255) / 256, 256, 0, stream>>>(W2, Wt2, K_DIM);
  gcn_prep_w3_kernel<<<(48 * K_DIM + 255) / 256, 256, 0, stream>>>(W3, b3, W3p, b3p, Wt3);

  // ---- layer 1: spmm(x)->hA ; relu(hA@W1+b1)->hB ; BN -> hA ----
  gcn_zero_kernel<<<(elemsH + 255) / 256, 256, 0, stream>>>(hA, elemsH);
  gcn_spmm_kernel<<<(spmm_thr + 255) / 256, 256, 0, stream>>>(erow, ecol, eval, x, hA, E);
  gcn_gemm_wmma_kernel<<<M / 16, 256, 0, stream>>>(hA, W1, Wt1, b1, hB, K_DIM, 1);
  gcn_zero_kernel<<<1, 256, 0, stream>>>(stats, 256);
  gcn_bn_stats_kernel<<<(M + 63) / 64, 128, 0, stream>>>(hB, stats, M);
  gcn_bn_norm_kernel<<<(elemsH + 255) / 256, 256, 0, stream>>>(hB, stats, gamma2, beta2, hA, M);

  // ---- layer 2: spmm(hA)->hB ; relu(hB@W2+b2)->hA ; BN -> hB ----
  gcn_zero_kernel<<<(elemsH + 255) / 256, 256, 0, stream>>>(hB, elemsH);
  gcn_spmm_kernel<<<(spmm_thr + 255) / 256, 256, 0, stream>>>(erow, ecol, eval, hA, hB, E);
  gcn_gemm_wmma_kernel<<<M / 16, 256, 0, stream>>>(hB, W2, Wt2, b2, hA, K_DIM, 1);
  gcn_zero_kernel<<<1, 256, 0, stream>>>(stats, 256);
  gcn_bn_stats_kernel<<<(M + 63) / 64, 128, 0, stream>>>(hA, stats, M);
  gcn_bn_norm_kernel<<<(elemsH + 255) / 256, 256, 0, stream>>>(hA, stats, gamma3, beta3, hB, M);

  // ---- layer 3: spmm(hB)->hA ; hA@W3p+b3p -> logits ; log_softmax -> out ----
  gcn_zero_kernel<<<(elemsH + 255) / 256, 256, 0, stream>>>(hA, elemsH);
  gcn_spmm_kernel<<<(spmm_thr + 255) / 256, 256, 0, stream>>>(erow, ecol, eval, hB, hA, E);
  gcn_gemm_wmma_kernel<<<M / 16, 96, 0, stream>>>(hA, W3p, Wt3, b3p, logits, 48, 0);
  gcn_logsoftmax_kernel<<<(M + 255) / 256, 256, 0, stream>>>(logits, out, M);
}
